// LoRALinear_52819507806382
// MI455X (gfx1250) — matrix operand order repaired and checked by reading, not compiled, not run on hardware
//
#include <hip/hip_runtime.h>

typedef __bf16 bf16_t;
typedef __attribute__((ext_vector_type(16))) __bf16 v16bf;
typedef __attribute__((ext_vector_type(4)))  __bf16 v4bf;
typedef __attribute__((ext_vector_type(8)))  float  v8f;
typedef __attribute__((ext_vector_type(4)))  unsigned int u32x4;
typedef __attribute__((ext_vector_type(8)))  int          i32x8;
typedef __attribute__((ext_vector_type(4)))  int          i32x4;

#define D_IN   4096
#define D_OUT  4096
#define M_TOT  8192
#define RANK   16
#define SCALE  2.0f   /* alpha/rank = 32/16 */

#define BM 128
#define BN 128
#define BK 32
#define NBUF 3

// ---------------------------------------------------------------------------
// Kernel 1: W_eff = W + (alpha/rank) * B @ A  ->  split into bf16 hi/lo planes
// ---------------------------------------------------------------------------
__global__ __launch_bounds__(256) void prep_weff(
    const float* __restrict__ W, const float* __restrict__ B,
    const float* __restrict__ A, bf16_t* __restrict__ whi,
    bf16_t* __restrict__ wlo)
{
    __shared__ float sB[64][RANK + 1];
    __shared__ float sA[RANK][64];
    const int ob = blockIdx.y * 64;
    const int ib = blockIdx.x * 64;
    const int t  = threadIdx.x;

    for (int j = t; j < 64 * RANK; j += 256) {
        int o = j >> 4, r = j & 15;
        sB[o][r] = B[(size_t)(ob + o) * RANK + r];
    }
    for (int j = t; j < RANK * 64; j += 256) {
        int r = j >> 6, i = j & 63;
        sA[r][i] = A[(size_t)r * D_IN + ib + i];
    }
    __syncthreads();

    for (int j = t; j < 64 * 64; j += 256) {
        int o = j >> 6, i = j & 63;
        float acc = 0.0f;
#pragma unroll
        for (int r = 0; r < RANK; ++r) acc += sB[o][r] * sA[r][i];
        size_t idx = (size_t)(ob + o) * D_IN + ib + i;
        float v  = W[idx] + SCALE * acc;
        bf16_t h = (bf16_t)v;
        float  l = v - (float)h;
        whi[idx] = h;
        wlo[idx] = (bf16_t)l;
    }
}

// ---------------------------------------------------------------------------
// Kernel 2: X (f32) -> bf16 hi/lo planes (one shot, removes in-GEMM cvt VALU)
// ---------------------------------------------------------------------------
__global__ __launch_bounds__(256) void prep_x(const float* __restrict__ X,
                                              bf16_t* __restrict__ xhi,
                                              bf16_t* __restrict__ xlo)
{
    const size_t i4 = ((size_t)blockIdx.x * 256 + threadIdx.x) * 4;
    const float4 v  = *(const float4*)(X + i4);
    bf16_t h0 = (bf16_t)v.x, h1 = (bf16_t)v.y, h2 = (bf16_t)v.z, h3 = (bf16_t)v.w;
    v4bf hi = {h0, h1, h2, h3};
    v4bf lo = {(bf16_t)(v.x - (float)h0), (bf16_t)(v.y - (float)h1),
               (bf16_t)(v.z - (float)h2), (bf16_t)(v.w - (float)h3)};
    *(v4bf*)(xhi + i4) = hi;
    *(v4bf*)(xlo + i4) = lo;
}

// ---------------------------------------------------------------------------
// Kernel 3: cbias = bias + (alpha/rank) * delta_bias
// ---------------------------------------------------------------------------
__global__ void prep_bias(const float* __restrict__ bias,
                          const float* __restrict__ dbias,
                          float* __restrict__ cbias)
{
    int i = blockIdx.x * blockDim.x + threadIdx.x;
    if (i < D_OUT) cbias[i] = bias[i] + SCALE * dbias[i];
}

// ---------------------------------------------------------------------------
// TDM descriptors (ISA 8.3/8.4): 2D tile [128 rows x 32 cols] bf16 from a
// row-major [tensor_rows x 4096] bf16 tensor. group1 is loop-invariant.
// ---------------------------------------------------------------------------
__device__ __forceinline__ i32x8 tdm_make_g1(unsigned tensor_rows)
{
    i32x8 g1;
    g1[0] = (int)(1u << 16);                                // wg_mask=0, data_size=1 (2B)
    g1[1] = (int)(((unsigned)D_IN & 0xFFFFu) << 16);        // tensor_dim0 lo16
    g1[2] = (int)(((unsigned)D_IN >> 16)                    // tensor_dim0 hi16
                  | ((tensor_rows & 0xFFFFu) << 16));       // tensor_dim1 lo16
    g1[3] = (int)((tensor_rows >> 16)                       // tensor_dim1 hi16
                  | ((unsigned)BK << 16));                  // tile_dim0 = 32
    g1[4] = (int)BM;                                        // tile_dim1 = 128, tile_dim2=0
    g1[5] = (int)D_IN;                                      // tensor_dim0_stride lo32
    g1[6] = 0;
    g1[7] = 0;
    return g1;
}

__device__ __forceinline__ void tdm_issue(const i32x8& g1, unsigned lds_off,
                                          const bf16_t* p)
{
    const unsigned long long ga = (unsigned long long)(uintptr_t)p;
    u32x4 g0;
    g0[0] = 1u;                                             // count=1 (valid), user mode
    g0[1] = lds_off;                                        // lds_addr[31:0]
    g0[2] = (unsigned int)ga;                               // global_addr[31:0]
    g0[3] = ((unsigned int)(ga >> 32) & 0x1FFFFFFu)         // global_addr[56:32]
            | (2u << 30);                                   // type=2 ("image")
    i32x4 z4 = {0, 0, 0, 0};
    i32x8 z8 = {0, 0, 0, 0, 0, 0, 0, 0};
    __builtin_amdgcn_tensor_load_to_lds(g0, g1, z4, z4, z8, 0);
}

// ---------------------------------------------------------------------------
// Fragment loaders matching CDNA5 bf16 WMMA VGPR layouts (ISA 7.12.2).
// ---------------------------------------------------------------------------
__device__ __forceinline__ v16bf load_a_frag(const bf16_t (*s)[BK], int rowBase, int lane)
{
    const int r  = rowBase + (lane & 15);
    const int kb = (lane >= 16) ? 8 : 0;
    union { uint4 u[2]; v16bf v; } cv;
    cv.u[0] = *(const uint4*)&s[r][kb];
    cv.u[1] = *(const uint4*)&s[r][kb + 16];
    return cv.v;
}

__device__ __forceinline__ v16bf load_b_frag(const bf16_t (*s)[BK], int nBase, int lane)
{
    const int n  = nBase + (lane & 15);
    const int kb = (lane >= 16) ? 8 : 0;
    const uint4* p = (const uint4*)&s[n][kb];
    union { uint4 u[2]; v16bf v; } cv;
    cv.u[0] = p[0];
    cv.u[1] = p[2];
    return cv.v;
}

__device__ __forceinline__ v8f wmma_bf16(v16bf a, v16bf b, v8f c)
{
    return __builtin_amdgcn_wmma_f32_16x16x32_bf16(
        false, a, false, b, (short)0, c, false, false);
}

// ---------------------------------------------------------------------------
// Kernel 4: out = X @ W_eff^T + cbias, split-bf16 3-product f32 emulation.
// 4 waves (wave32), each computing a 64x64 sub-tile (16 v8f accumulators) and
// each owning exactly one TDM stream. Triple-buffered LDS, issue-distance 2,
// one barrier per K-step, 48 WMMAs per barrier.
// ---------------------------------------------------------------------------
__global__ __launch_bounds__(128) void lora_gemm(
    const bf16_t* __restrict__ Xhi, const bf16_t* __restrict__ Xlo,
    const bf16_t* __restrict__ Whi, const bf16_t* __restrict__ Wlo,
    const float* __restrict__ cbias, float* __restrict__ Out)
{
    __shared__ __align__(16) bf16_t sXhi[NBUF][BM][BK];
    __shared__ __align__(16) bf16_t sXlo[NBUF][BM][BK];
    __shared__ __align__(16) bf16_t sWhi[NBUF][BN][BK];
    __shared__ __align__(16) bf16_t sWlo[NBUF][BN][BK];

    const int tid   = threadIdx.x;
    const int lane  = tid & 31;
    const int wave  = tid >> 5;     // 0..3
    const int waveM = wave >> 1;    // 0..1 : 64-row half
    const int waveN = wave & 1;     // 0..1 : 64-col half

    const int mb = blockIdx.y * BM;
    const int nb = blockIdx.x * BN;

    // ---- per-wave TDM stream (selected once; wave-uniform) ----
    const bf16_t* srcBase;
    unsigned rows, ldsBase;
    if (wave == 0)      { srcBase = Xhi + (size_t)mb * D_IN; rows = M_TOT;
                          ldsBase = (unsigned)(uintptr_t)&sXhi[0][0][0]; }
    else if (wave == 1) { srcBase = Xlo + (size_t)mb * D_IN; rows = M_TOT;
                          ldsBase = (unsigned)(uintptr_t)&sXlo[0][0][0]; }
    else if (wave == 2) { srcBase = Whi + (size_t)nb * D_IN; rows = D_OUT;
                          ldsBase = (unsigned)(uintptr_t)&sWhi[0][0][0]; }
    else                { srcBase = Wlo + (size_t)nb * D_IN; rows = D_OUT;
                          ldsBase = (unsigned)(uintptr_t)&sWlo[0][0][0]; }
    const i32x8 g1 = tdm_make_g1(rows);
    const unsigned bufStride = BM * BK * sizeof(bf16_t);    // 8 KB

    v8f acc[4][4];
#pragma unroll
    for (int mt = 0; mt < 4; ++mt)
#pragma unroll
        for (int nt = 0; nt < 4; ++nt)
#pragma unroll
            for (int j = 0; j < 8; ++j) acc[mt][nt][j] = 0.0f;

    // prime the pipeline: tiles for ki=0 and ki=1
    tdm_issue(g1, ldsBase + 0u * bufStride, srcBase);
    tdm_issue(g1, ldsBase + 1u * bufStride, srcBase + BK);

    const int NK = D_IN / BK; // 128
    for (int ki = 0; ki < NK; ++ki) {
        // retire own tile ki (TDM completes in order per wave), publish to all
        if (ki + 1 < NK) __builtin_amdgcn_s_wait_tensorcnt(1);
        else             __builtin_amdgcn_s_wait_tensorcnt(0);
        __syncthreads();

        // refill buffer last consumed at ki-1 (barrier above orders it)
        if (ki + 2 < NK)
            tdm_issue(g1, ldsBase + (unsigned)((ki + 2) % NBUF) * bufStride,
                      srcBase + (size_t)(ki + 2) * BK);

        const int cur = ki % NBUF;
        v16bf ahi[4], alo[4];
#pragma unroll
        for (int mt = 0; mt < 4; ++mt) {
            ahi[mt] = load_a_frag(sXhi[cur], waveM * 64 + mt * 16, lane);
            alo[mt] = load_a_frag(sXlo[cur], waveM * 64 + mt * 16, lane);
        }
#pragma unroll
        for (int nt = 0; nt < 4; ++nt) {
            v16bf bhi = load_b_frag(sWhi[cur], waveN * 64 + nt * 16, lane);
            v16bf blo = load_b_frag(sWlo[cur], waveN * 64 + nt * 16, lane);
#pragma unroll
            for (int mt = 0; mt < 4; ++mt) {
                acc[mt][nt] = wmma_bf16(ahi[mt], bhi, acc[mt][nt]); // hi*hi
                acc[mt][nt] = wmma_bf16(ahi[mt], blo, acc[mt][nt]); // hi*lo
                acc[mt][nt] = wmma_bf16(alo[mt], bhi, acc[mt][nt]); // lo*hi
            }
        }
        // no trailing barrier: NBUF=3 guarantees an intervening barrier
        // between the last readers of a buffer and its next TDM refill
    }

    // ---- epilogue: C/D layout (VGPR j: lanes0-15 -> M=j, lanes16-31 -> M=8+j)
    const int mrow0 = mb + waveM * 64;
    const int ncol0 = nb + waveN * 64;
#pragma unroll
    for (int mt = 0; mt < 4; ++mt) {
#pragma unroll
        for (int nt = 0; nt < 4; ++nt) {
            const int n  = ncol0 + nt * 16 + (lane & 15);
            const float bv = cbias[n];
            const int m0 = mrow0 + mt * 16 + (lane >> 4) * 8;
#pragma unroll
            for (int j = 0; j < 8; ++j) {
                Out[(size_t)(m0 + j) * D_OUT + n] = acc[mt][nt][j] + bv;
            }
        }
    }
}

// ---------------------------------------------------------------------------
extern "C" void kernel_launch(void* const* d_in, const int* in_sizes, int n_in,
                              void* d_out, int out_size, void* d_ws, size_t ws_size,
                              hipStream_t stream)
{
    const float* x     = (const float*)d_in[0];
    const float* W     = (const float*)d_in[1];
    const float* bias  = (const float*)d_in[2];
    const float* Blora = (const float*)d_in[3];
    const float* Alora = (const float*)d_in[4];
    const float* dbias = (const float*)d_in[5];
    float* out = (float*)d_out;

    char* ws = (char*)d_ws;
    const size_t wbytes = (size_t)D_OUT * D_IN * 2;  // 32 MB per W plane
    const size_t xbytes = (size_t)M_TOT * D_IN * 2;  // 64 MB per X plane
    bf16_t* whi  = (bf16_t*)ws;
    bf16_t* wlo  = (bf16_t*)(ws + wbytes);
    bf16_t* xhi  = (bf16_t*)(ws + 2 * wbytes);
    bf16_t* xlo  = (bf16_t*)(ws + 2 * wbytes + xbytes);
    float*  cbia = (float*)(ws + 2 * wbytes + 2 * xbytes);

    prep_weff<<<dim3(D_IN / 64, D_OUT / 64), 256, 0, stream>>>(W, Blora, Alora, whi, wlo);
    prep_x<<<(int)(((size_t)M_TOT * D_IN) / (256 * 4)), 256, 0, stream>>>(x, xhi, xlo);
    prep_bias<<<(D_OUT + 255) / 256, 256, 0, stream>>>(bias, dbias, cbia);
    lora_gemm<<<dim3(D_OUT / BN, M_TOT / BM), 128, 0, stream>>>(xhi, xlo, whi, wlo, cbia, out);
}
